// BinaryLinear_10325101380264
// MI455X (gfx1250) — matrix operand order repaired and checked by reading, not compiled, not run on hardware
//
#include <hip/hip_runtime.h>

// ---------------------------------------------------------------------------
// BinaryLinear: y = x @ sign(W)^T + bias   (8192x4096 @ 4096x4096)
// MI455X strategy:
//   * sign(W) in {-1,0,+1} is EXACT in bf16 -> V_WMMA_F32_16X16X32_BF16,
//     f32 accumulate. x split into bf16 hi+lo for near-fp32 accuracy
//     (2 WMMAs per tile into one accumulator).
//   * A slab (64 rows x 32 K, hi+lo) staged into LDS ONCE per block by the
//     Tensor Data Mover (tensor_load_to_lds), double-buffered, synced with
//     TENSORcnt + workgroup barriers. Waves read A via ds_load_b128.
//   * B = pre-swizzled sign(W) tiles streamed from global (each wave owns
//     distinct N columns, so no cross-wave reuse to exploit).
// ---------------------------------------------------------------------------

typedef __attribute__((ext_vector_type(16))) __bf16 v16bf;
typedef __attribute__((ext_vector_type(8)))  __bf16 v8bf;
typedef __attribute__((ext_vector_type(8)))  float  v8f;
typedef __attribute__((ext_vector_type(4)))  unsigned int v4u;
typedef __attribute__((ext_vector_type(8)))  int v8i;
typedef __attribute__((ext_vector_type(4)))  int v4i;

#define K_DIM 4096   // in_features
#define N_DIM 4096   // out_features
#define KT_TILES (K_DIM / 32)   // 128 K-tiles of 32
#define B_TILE_ELEMS 512        // 32x16 bf16 per B tile

__device__ __forceinline__ unsigned short f2bf_rne(float f) {
  unsigned int u = __float_as_uint(f);
  unsigned int r = u + 0x7FFFu + ((u >> 16) & 1u);   // round-to-nearest-even
  return (unsigned short)(r >> 16);
}
__device__ __forceinline__ float bf2f(unsigned short h) {
  return __uint_as_float(((unsigned int)h) << 16);
}

// ---------------------------------------------------------------------------
// Kernel 1: binarize W -> bf16 {-1,0,+1}, pre-swizzled in this file's contiguous-k B-operand
// (32x16 KxN) per-lane layout (lanes 0-15: N=lane, K=0..15; lanes 16-31:
// N=lane-16, K=16..31), tiles ordered [nt][kt] for contiguous K streaming.
// ---------------------------------------------------------------------------
__global__ void quant_w_kernel(const float* __restrict__ w,
                               unsigned short* __restrict__ wb) {
  size_t t = (size_t)blockIdx.x * blockDim.x + threadIdx.x;  // 0..16M-1
  int n = (int)(t >> 12);      // out_feature = GEMM N
  int k = (int)(t & 4095);     // in_feature  = GEMM K
  float v = w[t];
  unsigned short s = (v > 0.0f) ? 0x3F80u : ((v < 0.0f) ? 0xBF80u : 0u);

  int nt  = n >> 4;
  int kt  = k >> 5;
  int kin = k & 31;
  int lane = (n & 15) + ((kin & 16) ? 16 : 0);
  int j    = kin & 15;
  wb[((size_t)(nt * KT_TILES + kt) << 9) + lane * 16 + j] = s;
}

// ---------------------------------------------------------------------------
// Kernel 2: x -> bf16 hi + lo residual, row-major.
// ---------------------------------------------------------------------------
__global__ void split_x_kernel(const float* __restrict__ x,
                               unsigned short* __restrict__ xh,
                               unsigned short* __restrict__ xl,
                               size_t nelem) {
  size_t t = (size_t)blockIdx.x * blockDim.x + threadIdx.x;
  if (t >= nelem) return;
  float f = x[t];
  unsigned short hi = f2bf_rne(f);
  unsigned short lo = f2bf_rne(f - bf2f(hi));
  xh[t] = hi;
  xl[t] = lo;
}

// ---------------------------------------------------------------------------
// TDM descriptor build + issue for one 64-row x 32-col bf16 tile of a
// row-major [8192][4096] tensor, landed row-major into LDS (row stride 64B).
// D# bitfields per CDNA5 ISA ch.8 (group0: ctrl/lds_addr/global_addr/type,
// group1: data_size, dims, tile dims, strides). This toolchain's builtin is
// the 6-arg form: (g0, g1, g2, g3, extra_v8i, cpol).
// ---------------------------------------------------------------------------
__device__ __forceinline__ void tdm_load_tile(unsigned lds_byte_addr,
                                              unsigned long long gaddr) {
  v4u g0;
  g0[0] = 1u;                                  // count=1, user descriptor
  g0[1] = lds_byte_addr;                       // lds_addr
  g0[2] = (unsigned)(gaddr & 0xFFFFFFFFu);     // global_addr[31:0]
  g0[3] = (unsigned)((gaddr >> 32) & 0x1FFFFFFu) | 0x80000000u; // [56:32] | type=2

  v8i g1;
  g1[0] = 0x00010000;          // workgroup_mask=0, data_size=1 (2 bytes)
  g1[1] = (int)(((unsigned)K_DIM & 0xFFFFu) << 16);  // tensor_dim0[15:0]=4096
  g1[2] = (int)(((unsigned)8192 & 0xFFFFu) << 16);   // dim0[31:16]=0, dim1[15:0]=8192
  g1[3] = (int)(32u << 16);    // dim1[31:16]=0, tile_dim0=32
  g1[4] = 64;                  // tile_dim1=64, tile_dim2=0
  g1[5] = K_DIM;               // tensor_dim0_stride[31:0]=4096
  g1[6] = 0;                   // dim0_stride[47:32]=0, dim1_stride[15:0]=0
  g1[7] = 0;                   // dim1_stride[47:16]=0

  v4i gz4 = {0, 0, 0, 0};                  // groups 2/3 unused (2D tensor)
  v8i gz8 = {0, 0, 0, 0, 0, 0, 0, 0};      // trailing descriptor group: unused

  __builtin_amdgcn_tensor_load_to_lds(g0, g1, gz4, gz4, gz8, 0);
}

// ---------------------------------------------------------------------------
// Kernel 3: GEMM. Block = 256 threads = 8 waves; block tile = 64M x 128N.
// Wave w owns N columns [w*16, w*16+16). A staged via TDM (double buffered).
// ---------------------------------------------------------------------------
__global__ void __launch_bounds__(256)
gemm_bf16_split_kernel(const unsigned short* __restrict__ xh,
                       const unsigned short* __restrict__ xl,
                       const unsigned short* __restrict__ wb,
                       const float* __restrict__ bias,
                       float* __restrict__ out) {
  // [stage][hi/lo][row*32+col] bf16, row-major 64x32, row stride 64B. 16 KB.
  __shared__ unsigned short smA[2][2][64 * 32];

  const int lane  = threadIdx.x & 31;
  const int wave  = threadIdx.x >> 5;
  const int laneN = lane & 15;
  const int hi16  = lane >> 4;                   // 0 / 1

  const int n0 = blockIdx.x * 128 + wave * 16;
  const int m0 = blockIdx.y * 64;

  v8f acc[4];
#pragma unroll
  for (int mt = 0; mt < 4; ++mt) acc[mt] = (v8f){};

  const unsigned short* bptr =
      wb + (size_t)(n0 >> 4) * (KT_TILES * B_TILE_ELEMS) + lane * 16;

  const unsigned long long gxh = (unsigned long long)(size_t)xh;
  const unsigned long long gxl = (unsigned long long)(size_t)xl;
  const size_t mrow = (size_t)m0 * K_DIM;

  // Prologue: stage 0 in flight.
  if (wave == 0) {
    tdm_load_tile((unsigned)(size_t)&smA[0][0][0], gxh + 2ull * mrow);
    tdm_load_tile((unsigned)(size_t)&smA[0][1][0], gxl + 2ull * mrow);
  }

  // Per-lane A fragment offsets inside a staged 64x32 tile (elements):
  //   row = mt*16 + laneN, kofs = (hi16?16:0) + half*8
  const int aofs = laneN * 32 + hi16 * 16;       // + mt*512 + half*8

  for (int kt = 0; kt < KT_TILES; ++kt) {
    const int cur = kt & 1;
    if (kt + 1 < KT_TILES) {
      if (wave == 0) {
        const size_t off = 2ull * (mrow + (size_t)(kt + 1) * 32);
        tdm_load_tile((unsigned)(size_t)&smA[cur ^ 1][0][0], gxh + off);
        tdm_load_tile((unsigned)(size_t)&smA[cur ^ 1][1][0], gxl + off);
        __builtin_amdgcn_s_wait_tensorcnt(2);    // stage kt complete
      }
    } else if (wave == 0) {
      __builtin_amdgcn_s_wait_tensorcnt(0);
    }
    __syncthreads();                             // publish buf[cur]

    union { v16bf v; v8bf h[2]; } b;
    b.h[0] = *(const v8bf*)(bptr);
    b.h[1] = *(const v8bf*)(bptr + 8);
    bptr += B_TILE_ELEMS;
    __builtin_prefetch(bptr, 0, 1);              // next B tile

#pragma unroll
    for (int mt = 0; mt < 4; ++mt) {
      const unsigned short* ah_p = &smA[cur][0][mt * 512 + aofs];
      const unsigned short* al_p = &smA[cur][1][mt * 512 + aofs];
      union { v16bf v; v8bf h[2]; } ah, al;
      ah.h[0] = *(const v8bf*)(ah_p);
      ah.h[1] = *(const v8bf*)(ah_p + 8);
      al.h[0] = *(const v8bf*)(al_p);
      al.h[1] = *(const v8bf*)(al_p + 8);

      acc[mt] = __builtin_amdgcn_wmma_f32_16x16x32_bf16(
          false, ah.v, false, b.v, (short)0, acc[mt], false, false);
      acc[mt] = __builtin_amdgcn_wmma_f32_16x16x32_bf16(
          false, al.v, false, b.v, (short)0, acc[mt], false, false);
    }
    __syncthreads();                             // done reading buf[cur]
  }

  // Epilogue. C/D layout: VGPR r -> row (r + 8*hi16), lane%16 -> col.
  const int   col = n0 + laneN;
  const float bv  = bias[col];
#pragma unroll
  for (int mt = 0; mt < 4; ++mt) {
#pragma unroll
    for (int r = 0; r < 8; ++r) {
      const int row = m0 + mt * 16 + r + hi16 * 8;
      out[(size_t)row * N_DIM + col] = acc[mt][r] + bv;
    }
  }
}

// ---------------------------------------------------------------------------
extern "C" void kernel_launch(void* const* d_in, const int* in_sizes, int n_in,
                              void* d_out, int out_size, void* d_ws, size_t ws_size,
                              hipStream_t stream) {
  const float* x    = (const float*)d_in[0];
  const float* w    = (const float*)d_in[1];
  const float* bias = (const float*)d_in[2];
  float*       out  = (float*)d_out;

  const int batch = in_sizes[0] / K_DIM;          // 8192
  const size_t nx = (size_t)batch * K_DIM;

  unsigned short* xh = (unsigned short*)d_ws;     // 64 MB
  unsigned short* xl = xh + nx;                   // 64 MB
  unsigned short* wb = xl + nx;                   // 32 MB

  split_x_kernel<<<(unsigned)((nx + 255) / 256), 256, 0, stream>>>(x, xh, xl, nx);

  const size_t nw = (size_t)N_DIM * K_DIM;
  quant_w_kernel<<<(unsigned)(nw / 256), 256, 0, stream>>>(w, wb);

  dim3 grid(N_DIM / 128, batch / 64);             // (32, 128)
  gemm_bf16_split_kernel<<<grid, 256, 0, stream>>>(xh, xl, wb, bias, out);
}
